// ExpertModule_trm_15668040696090
// MI455X (gfx1250) — compile-verified, matrix operand order checked by reading, not run on hardware
//
#include <hip/hip_runtime.h>

// ---------------------------------------------------------------------------
// Types / constants
// ---------------------------------------------------------------------------
typedef __attribute__((ext_vector_type(16))) _Float16 v16h;
typedef __attribute__((ext_vector_type(8)))  float    v8f;

#define B_    16
#define S_    2048
#define D_    400
#define HS_   100
#define DP_   416    // D padded to 13*32
#define HP_   128    // head padded to 4*32
#define NQK_  256    // Q (cols 0..127) and K (cols 128..255) row-major stride
#define NALL_ 384    // Q|K|V projection output width (3*128)
// SCALE * log2(e) = 0.1 * 1.4426950408889634
#define F_L2E 0.14426950408889634f

union Frag16 { v16h h; unsigned u[8]; };

__device__ __forceinline__ int kbase(int vv, int hi) {
  // K index of element pair {2*vv, 2*vv+1} per CDNA5 16-bit A/B fragment layout
  return ((vv & 4) << 2) + ((vv & 3) << 1) + (hi << 3);
}

__device__ __forceinline__ v16h load_frag(const _Float16* rowptr, int hi) {
  // rowptr already includes (row * ld); lowers to two global_load_b128
  Frag16 f;
#pragma unroll
  for (int vv = 0; vv < 8; ++vv)
    f.u[vv] = *(const unsigned*)(rowptr + kbase(vv, hi));
  return f.h;
}

__device__ __forceinline__ v8f vzero() {
  v8f z;
#pragma unroll
  for (int i = 0; i < 8; ++i) z[i] = 0.f;
  return z;
}

__device__ __forceinline__ v8f wmma_f16(v16h a, v16h b, v8f c) {
  return __builtin_amdgcn_wmma_f32_16x16x32_f16(false, a, false, b, (short)0, c,
                                                false, false);
}

// ---------------------------------------------------------------------------
// Prep kernels
// ---------------------------------------------------------------------------
__global__ void cvt_x_kernel(const float* __restrict__ x, _Float16* __restrict__ xh) {
  long i = (long)blockIdx.x * 256 + threadIdx.x;
  if (i >= (long)B_ * S_ * DP_) return;
  int r = (int)(i / DP_), c = (int)(i % DP_);
  float v = (c < D_) ? x[(long)r * D_ + c] : 0.f;
  xh[i] = (_Float16)v;
}

// WallT[n][k] = attn_kernel[n>>7][k][n&127] (zero-padded), stored [384,416]
__global__ void prep_wall_kernel(const float* __restrict__ ak, _Float16* __restrict__ wallT) {
  int i = blockIdx.x * 256 + threadIdx.x;
  if (i >= NALL_ * DP_) return;
  int n = i / DP_, k = i % DP_;
  int m = n >> 7, h = n & 127;
  float v = 0.f;
  if (k < D_ && h < HS_) v = ak[((long)m * D_ + k) * HS_ + h];
  wallT[i] = (_Float16)v;
}

// Wcomb[j,e] = sum_k (sum_h attn_dense[h*100+j, k]) * d1_w[k, e]; store transposed [400,128]
__global__ void prep_comb_kernel(const float* __restrict__ ad, const float* __restrict__ d1w,
                                 _Float16* __restrict__ wcT) {
  int i = blockIdx.x * 256 + threadIdx.x;
  if (i >= D_ * HP_) return;
  int e = i / HP_, j = i % HP_;
  float acc = 0.f;
  if (j < HS_) {
    for (int k = 0; k < D_; ++k) {
      float wad = ad[j * D_ + k] + ad[(HS_ + j) * D_ + k] +
                  ad[(2 * HS_ + j) * D_ + k] + ad[(3 * HS_ + j) * D_ + k];
      acc += wad * d1w[k * D_ + e];
    }
  }
  wcT[i] = (_Float16)acc;
}

__global__ void init_pool_kernel(unsigned* __restrict__ pmax, float* __restrict__ psum) {
  int i = blockIdx.x * 256 + threadIdx.x;
  if (i < B_ * D_) { pmax[i] = 0u; psum[i] = 0.f; }
}

// ---------------------------------------------------------------------------
// GEMM1: [32768,416] x [416,384] -> Q,K row-major [32768,256] + V transposed
// One wave per 16-row tile; A fragments cached in registers. All 13 B
// fragments preloaded as an array so the 26 b128 loads issue as clauses and
// the 13 WMMAs retire behind decreasing partial loadcnt waits.
// ---------------------------------------------------------------------------
__global__ __launch_bounds__(128) void gemm1_kernel(const _Float16* __restrict__ xh,
                                                    const _Float16* __restrict__ wallT,
                                                    _Float16* __restrict__ qk,
                                                    _Float16* __restrict__ vt) {
  int lane = threadIdx.x & 31, wave = threadIdx.x >> 5;
  int hi = lane >> 4, nn = lane & 15;
  int tile = blockIdx.x * 4 + wave;          // 0..2047
  long r0 = (long)tile * 16;
  v16h a[13];
#pragma unroll
  for (int c = 0; c < 13; ++c)
    a[c] = load_frag(xh + (r0 + nn) * DP_ + c * 32, hi);
  for (int nt = 0; nt < 24; ++nt) {
    const _Float16* brow = wallT + (long)(nt * 16 + nn) * DP_;
    v16h bw[13];
#pragma unroll
    for (int c = 0; c < 13; ++c)
      bw[c] = load_frag(brow + c * 32, hi);
    v8f acc = vzero();
#pragma unroll
    for (int c = 0; c < 13; ++c)
      acc = wmma_f16(a[c], bw[c], acc);
    int col = nt * 16 + nn;
    if (nt < 16) {                            // Q,K row-major
#pragma unroll
      for (int v = 0; v < 8; ++v)
        qk[(r0 + v + (hi << 3)) * NQK_ + col] = (_Float16)acc[v];
    } else {                                  // V transposed: [b][head][s]
      int head = col - 256;
#pragma unroll
      for (int v = 0; v < 8; ++v) {
        long tok = r0 + v + (hi << 3);
        int bb = (int)(tok >> 11), sr = (int)(tok & 2047);
        vt[((long)bb * HP_ + head) * S_ + sr] = (_Float16)acc[v];
      }
    }
  }
}

// ---------------------------------------------------------------------------
// Flash attention: one wave per 16-query tile, 32 keys / iteration.
// Online softmax with per-row m/l; P converted f32->f16 through LDS.
// K-tile fragments preloaded before score WMMAs; V-tile fragments hoisted
// above the softmax VALU block so exp/max math overlaps VMEM latency.
// ---------------------------------------------------------------------------
__global__ __launch_bounds__(128) void flash_kernel(const _Float16* __restrict__ qk,
                                                    const _Float16* __restrict__ vt,
                                                    _Float16* __restrict__ vo) {
  __shared__ _Float16 plds[4][16 * 32];
  int lane = threadIdx.x & 31, wave = threadIdx.x >> 5;
  int hi = lane >> 4, nn = lane & 15;
  int tile = blockIdx.x * 4 + wave;          // 0..2047
  int b = tile >> 7;
  int q0 = (tile & 127) << 4;
  long base = (long)b * S_ + q0;

  v16h aq[4];
#pragma unroll
  for (int c = 0; c < 4; ++c)
    aq[c] = load_frag(qk + (base + nn) * NQK_ + c * 32, hi);

  v8f o[8];
  float m[8], l[8];
#pragma unroll
  for (int t = 0; t < 8; ++t) o[t] = vzero();
#pragma unroll
  for (int v = 0; v < 8; ++v) { m[v] = -3.0e38f; l[v] = 0.f; }

  const _Float16* kptr  = qk + (long)b * S_ * NQK_ + 128;  // K at col offset 128
  const _Float16* vbase = vt + (long)b * HP_ * S_;
  _Float16* pl = plds[wave];

  for (int kt = 0; kt < S_ / 32; ++kt) {
    int key0 = kt * 32;

    // ---- scores: preload all 8 B fragments, then 8 WMMAs
    v16h bk[8];
#pragma unroll
    for (int c = 0; c < 4; ++c) {
      bk[2 * c]     = load_frag(kptr + (long)(key0 + nn) * NQK_ + c * 32, hi);
      bk[2 * c + 1] = load_frag(kptr + (long)(key0 + 16 + nn) * NQK_ + c * 32, hi);
    }
    v8f s0 = vzero(), s1 = vzero();
#pragma unroll
    for (int c = 0; c < 4; ++c) {
      s0 = wmma_f16(aq[c], bk[2 * c], s0);
      s1 = wmma_f16(aq[c], bk[2 * c + 1], s1);
    }

    // ---- V fragments issued early: latency overlaps softmax VALU below
    v16h bv[8];
#pragma unroll
    for (int t = 0; t < 8; ++t)
      bv[t] = load_frag(vbase + (long)(t * 16 + nn) * S_ + key0, hi);

    // ---- online softmax (per-row shfl reductions within 16-lane halves)
#pragma unroll
    for (int v = 0; v < 8; ++v) {
      float t = fmaxf(s0[v], s1[v]);
      t = fmaxf(t, __shfl_xor(t, 1, 32));
      t = fmaxf(t, __shfl_xor(t, 2, 32));
      t = fmaxf(t, __shfl_xor(t, 4, 32));
      t = fmaxf(t, __shfl_xor(t, 8, 32));
      float mn   = fmaxf(m[v], t);
      float corr = exp2f((m[v] - mn) * F_L2E);
      float p0   = exp2f((s0[v] - mn) * F_L2E);
      float p1   = exp2f((s1[v] - mn) * F_L2E);
      float r = p0 + p1;
      r += __shfl_xor(r, 1, 32);
      r += __shfl_xor(r, 2, 32);
      r += __shfl_xor(r, 4, 32);
      r += __shfl_xor(r, 8, 32);
      l[v] = l[v] * corr + r;
      m[v] = mn;
#pragma unroll
      for (int t2 = 0; t2 < 8; ++t2) o[t2][v] *= corr;
      int row = v + (hi << 3);
      pl[row * 32 + nn]      = (_Float16)p0;
      pl[row * 32 + 16 + nn] = (_Float16)p1;
    }
    __syncthreads();

    // ---- P@V: reload P in A layout from LDS, 8 accumulating WMMAs
    v16h ap = load_frag(pl + nn * 32, hi);
#pragma unroll
    for (int t = 0; t < 8; ++t)
      o[t] = wmma_f16(ap, bv[t], o[t]);
    __syncthreads();
  }

#pragma unroll
  for (int v = 0; v < 8; ++v) {
    float inv = 1.f / l[v];
    long gq = base + v + (hi << 3);
#pragma unroll
    for (int t = 0; t < 8; ++t)
      vo[gq * HP_ + t * 16 + nn] = (_Float16)(o[t][v] * inv);
  }
}

// ---------------------------------------------------------------------------
// GEMM2 (+fused relu/bias and max/mean pooling):
// H = relu(Vattn[32768,128] @ Wcomb[128,400] + d1_b); pool over S per batch.
// ---------------------------------------------------------------------------
__global__ __launch_bounds__(128) void gemm2_pool_kernel(const _Float16* __restrict__ va,
                                                         const _Float16* __restrict__ wcT,
                                                         const float* __restrict__ d1b,
                                                         unsigned* __restrict__ pmax,
                                                         float* __restrict__ psum) {
  int lane = threadIdx.x & 31, wave = threadIdx.x >> 5;
  int hi = lane >> 4, nn = lane & 15;
  int tile = blockIdx.x * 4 + wave;          // 0..2047
  long r0 = (long)tile * 16;
  int b = (int)(r0 >> 11);
  v16h a[4];
#pragma unroll
  for (int c = 0; c < 4; ++c)
    a[c] = load_frag(va + (r0 + nn) * HP_ + c * 32, hi);
  for (int nt = 0; nt < 25; ++nt) {
    v16h bw[4];
#pragma unroll
    for (int c = 0; c < 4; ++c)
      bw[c] = load_frag(wcT + (long)(nt * 16 + nn) * HP_ + c * 32, hi);
    v8f acc = vzero();
#pragma unroll
    for (int c = 0; c < 4; ++c)
      acc = wmma_f16(a[c], bw[c], acc);
    int col = nt * 16 + nn;
    float bias = d1b[col];
    float vmax = 0.f, vsum = 0.f;
#pragma unroll
    for (int v = 0; v < 8; ++v) {
      float h = fmaxf(acc[v] + bias, 0.f);   // relu
      vmax = fmaxf(vmax, h);
      vsum += h;
    }
    vmax = fmaxf(vmax, __shfl_xor(vmax, 16, 32));
    vsum += __shfl_xor(vsum, 16, 32);
    if (lane < 16) {                          // non-negative floats: uint max OK
      atomicMax(pmax + b * D_ + col, __float_as_uint(vmax));
      atomicAdd(psum + b * D_ + col, vsum);
    }
  }
}

// ---------------------------------------------------------------------------
// FC head: pooled[16,800] -> relu fc1[512] -> relu fc2[256]
// ---------------------------------------------------------------------------
__global__ __launch_bounds__(256) void fc_kernel(const unsigned* __restrict__ pmax,
                                                 const float* __restrict__ psum,
                                                 const float* __restrict__ w1,
                                                 const float* __restrict__ b1,
                                                 const float* __restrict__ w2,
                                                 const float* __restrict__ b2,
                                                 float* __restrict__ out) {
  __shared__ float p[800];
  __shared__ float h1[512];
  int b = blockIdx.x, t = threadIdx.x;
  for (int i = t; i < 800; i += 256)
    p[i] = (i < 400) ? __uint_as_float(pmax[b * D_ + i])
                     : psum[b * D_ + (i - 400)] * (1.f / (float)S_);
  __syncthreads();
  for (int j = t; j < 512; j += 256) {
    float acc = b1[j];
    for (int k = 0; k < 800; ++k) acc += p[k] * w1[k * 512 + j];
    h1[j] = fmaxf(acc, 0.f);
  }
  __syncthreads();
  {
    int j = t;
    float acc = b2[j];
    for (int k = 0; k < 512; ++k) acc += h1[k] * w2[k * 256 + j];
    out[b * 256 + j] = fmaxf(acc, 0.f);
  }
}

// ---------------------------------------------------------------------------
// Launch
// ---------------------------------------------------------------------------
extern "C" void kernel_launch(void* const* d_in, const int* in_sizes, int n_in,
                              void* d_out, int out_size, void* d_ws, size_t ws_size,
                              hipStream_t stream) {
  const float* x   = (const float*)d_in[0];
  const float* ak  = (const float*)d_in[1];
  const float* ad  = (const float*)d_in[2];
  const float* d1w = (const float*)d_in[3];
  const float* d1b = (const float*)d_in[4];
  const float* f1w = (const float*)d_in[5];
  const float* f1b = (const float*)d_in[6];
  const float* f2w = (const float*)d_in[7];
  const float* f2b = (const float*)d_in[8];
  float* out = (float*)d_out;

  char* ws = (char*)d_ws;
  _Float16* xh    = (_Float16*)(ws + 0);          // 32768*416*2 = 27,262,976
  _Float16* wallT = (_Float16*)(ws + 27262976);   //   384*416*2 =    319,488
  _Float16* qk    = (_Float16*)(ws + 27582464);   // 32768*256*2 = 16,777,216
  _Float16* vt    = (_Float16*)(ws + 44359680);   // 16*128*2048*2 = 8,388,608
  _Float16* va    = (_Float16*)(ws + 52748288);   // 32768*128*2 =  8,388,608
  _Float16* wcT   = (_Float16*)(ws + 61136896);   //   400*128*2 =    102,400
  unsigned* pmax  = (unsigned*)(ws + 61239296);   //   16*400*4  =     25,600
  float*    psum  = (float*)   (ws + 61264896);   //   16*400*4  =     25,600

  cvt_x_kernel<<<(B_ * S_ * DP_ + 255) / 256, 256, 0, stream>>>(x, xh);
  prep_wall_kernel<<<(NALL_ * DP_ + 255) / 256, 256, 0, stream>>>(ak, wallT);
  prep_comb_kernel<<<(D_ * HP_ + 255) / 256, 256, 0, stream>>>(ad, d1w, wcT);
  init_pool_kernel<<<(B_ * D_ + 255) / 256, 256, 0, stream>>>(pmax, psum);

  gemm1_kernel<<<512, 128, 0, stream>>>(xh, wallT, qk, vt);
  flash_kernel<<<512, 128, 0, stream>>>(qk, vt, va);
  gemm2_pool_kernel<<<512, 128, 0, stream>>>(va, wcT, d1b, pmax, psum);
  fc_kernel<<<B_, 256, 0, stream>>>(pmax, psum, f1w, f1b, f2w, f2b, out);
}
